// LSTMDynamicsModule_32152125178147
// MI455X (gfx1250) — compile-verified, matrix operand order checked by reading, not run on hardware
//
#include <hip/hip_runtime.h>
#include <hip/hip_bf16.h>

typedef __bf16 bf16_t;
typedef __attribute__((ext_vector_type(16))) __bf16 bf16x16;
typedef __attribute__((ext_vector_type(8)))  __bf16 bf16x8;
typedef __attribute__((ext_vector_type(8)))  float  f32x8;

// Problem sizes (fixed by reference)
constexpr int B_  = 1024;
constexpr int K_  = 16;
constexpr int H_  = 512;
constexpr int I_  = 512;
constexpr int CD_ = I_ + H_;          // fused contraction: x-part then hidden-part

// Tiling
constexpr int BM = 128;               // batch rows per workgroup
constexpr int BN = 64;                // h columns per gate per workgroup
constexpr int BK = 32;                // bf16 k-step (one wmma K)
constexpr int NIT = CD_ / BK;         // 32 K-steps
constexpr int A_STRIDE = BK + 8;      // 40 bf16 -> 80B row stride (16B aligned)
constexpr int B_STRIDE = BK + 8;
constexpr int BUF_ELEMS = BM * A_STRIDE + 4 * BN * B_STRIDE;   // 15360 bf16 per buffer

__device__ __forceinline__ float fast_sigmoid(float v) {
    return __builtin_amdgcn_rcpf(1.0f + __expf(-v));
}
__device__ __forceinline__ float fast_tanh(float v) {
    float a = __builtin_fabsf(v);
    float e = __expf(-2.0f * a);
    float r = (1.0f - e) * __builtin_amdgcn_rcpf(1.0f + e);
    return __builtin_copysignf(r, v);
}
__device__ __forceinline__ void cvt4(bf16x8& d, int o, float4 v) {
    d[o] = (bf16_t)v.x; d[o + 1] = (bf16_t)v.y; d[o + 2] = (bf16_t)v.z; d[o + 3] = (bf16_t)v.w;
}

__global__ __launch_bounds__(256)
void lstm_fused_wmma(
    const float* __restrict__ hidden, const float* __restrict__ x,
    const int*   __restrict__ mask,   const float* __restrict__ cell,
    const float* __restrict__ Wi0, const float* __restrict__ Wh0, const float* __restrict__ bb0,
    const float* __restrict__ Wi1, const float* __restrict__ Wh1, const float* __restrict__ bb1,
    const float* __restrict__ Wi2, const float* __restrict__ Wh2, const float* __restrict__ bb2,
    const float* __restrict__ Wi3, const float* __restrict__ Wh3, const float* __restrict__ bb3,
    float* __restrict__ out)
{
    __shared__ bf16_t smem[2 * BUF_ELEMS];   // ping-pong buffers (60 KB)

    const int k     = blockIdx.z;            // expert slice (slowest -> L2 weight residency)
    const int mTile = blockIdx.y * BM;
    const int nTile = blockIdx.x * BN;

    const int tid   = threadIdx.x;
    const int lane  = tid & 31;
    const int wave  = tid >> 5;
    const int waveM = (wave & 3) * 32;       // 4 waves along M
    const int waveN = (wave >> 2) * 32;      // 2 waves along N
    const int lrow  = lane & 15;
    const int lhi   = lane >> 4;

    const float* Wi[4]  = {Wi0, Wi1, Wi2, Wi3};
    const float* Wh[4]  = {Wh0, Wh1, Wh2, Wh3};
    const float* bsv[4] = {bb0, bb1, bb2, bb3};

    const f32x8 zero = {0.f,0.f,0.f,0.f,0.f,0.f,0.f,0.f};
    f32x8 acc[4][2][2];
    #pragma unroll
    for (int g = 0; g < 4; ++g)
        #pragma unroll
        for (int mi = 0; mi < 2; ++mi)
            #pragma unroll
            for (int ni = 0; ni < 2; ++ni)
                acc[g][mi][ni] = zero;

    // A staging: thread -> row = tid/2 (0..127), 16 floats at col (tid&1)*16
    const int aRow = tid >> 1;
    const int aCol = (tid & 1) * 16;
    // B staging (per gate): column strip -> n = tid&63, k-chunk = (tid>>6)*8
    const int bN     = tid & 63;
    const int bKBase = (tid >> 6) * 8;

    // Staged tile (registers), already converted to bf16
    bf16x8 sA0, sA1;
    bf16x8 sB[4];

    auto loadTile = [&](int kc0) {
        const bool xPart = (kc0 < I_);
        const int  kcL   = kc0 & (I_ - 1);
        {
            const float* src = xPart ? x : hidden;     // both [B,K,512]
            const float* p = src + (size_t)(mTile + aRow) * (K_ * I_) + (size_t)k * I_ + kcL + aCol;
            cvt4(sA0, 0, *(const float4*)(p + 0));
            cvt4(sA0, 4, *(const float4*)(p + 4));
            cvt4(sA1, 0, *(const float4*)(p + 8));
            cvt4(sA1, 4, *(const float4*)(p + 12));
        }
        #pragma unroll
        for (int g = 0; g < 4; ++g) {
            const float* wsrc = xPart ? Wi[g] : Wh[g]; // both [K,512,512]
            const float* p = wsrc + (size_t)k * (I_ * H_) + (size_t)(kcL + bKBase) * H_ + nTile + bN;
            #pragma unroll
            for (int j = 0; j < 8; ++j)
                sB[g][j] = (bf16_t)p[(size_t)j * H_];  // coalesced across lanes (consecutive n)
        }
    };

    auto storeTile = [&](bf16_t* buf) {
        bf16_t* dA = buf + aRow * A_STRIDE + aCol;
        *(bf16x8*)(dA + 0) = sA0;
        *(bf16x8*)(dA + 8) = sA1;
        bf16_t* bufB = buf + BM * A_STRIDE;            // transposed [n][k] layout
        #pragma unroll
        for (int g = 0; g < 4; ++g)
            *(bf16x8*)(bufB + (size_t)(g * BN + bN) * B_STRIDE + bKBase) = sB[g];
    };

    auto computeFrom = [&](const bf16_t* buf) {
        const bf16_t* bufA = buf;
        const bf16_t* bufB = buf + BM * A_STRIDE;
        // A/B 16-bit lane layout: lanes 0-15 hold K{0..7,16..23}, lanes 16-31 hold K{8..15,24..31}
        bf16x16 aF[2];
        #pragma unroll
        for (int mi = 0; mi < 2; ++mi) {
            const bf16_t* base = bufA + (waveM + mi * 16 + lrow) * A_STRIDE;
            union { bf16x16 v; bf16x8 h[2]; } u;
            u.h[0] = *(const bf16x8*)(base + lhi * 8);
            u.h[1] = *(const bf16x8*)(base + 16 + lhi * 8);
            aF[mi] = u.v;
        }
        #pragma unroll
        for (int g = 0; g < 4; ++g) {
            #pragma unroll
            for (int ni = 0; ni < 2; ++ni) {
                const bf16_t* base = bufB + (g * BN + waveN + ni * 16 + lrow) * B_STRIDE;
                union { bf16x16 v; bf16x8 h[2]; } u;
                u.h[0] = *(const bf16x8*)(base + lhi * 8);
                u.h[1] = *(const bf16x8*)(base + 16 + lhi * 8);
                const bf16x16 bF = u.v;
                #pragma unroll
                for (int mi = 0; mi < 2; ++mi) {
                    acc[g][mi][ni] = __builtin_amdgcn_wmma_f32_16x16x32_bf16(
                        false, aF[mi], false, bF, (short)0, acc[g][mi][ni], false, false);
                }
            }
        }
    };

    // ---- pipelined main loop (single barrier per K-step) ----
    loadTile(0);
    storeTile(smem);          // buffer 0
    int cur = 0;
    for (int it = 0; it < NIT; ++it) {
        if (it + 1 < NIT) loadTile((it + 1) * BK);
        if (it + 2 < NIT) {   // prefetch 2 steps ahead (global_prefetch_b8)
            const int  kcp = (it + 2) * BK;
            const bool xp  = kcp < I_;
            const int  kl  = kcp & (I_ - 1);
            const float* ps = xp ? x : hidden;
            __builtin_prefetch(ps + (size_t)(mTile + aRow) * (K_ * I_) + (size_t)k * I_ + kl + aCol, 0, 0);
            const float* pw = (xp ? Wi[0] : Wh[0]) +
                (size_t)k * (I_ * H_) + (size_t)(kl + bKBase) * H_ + nTile + bN;
            __builtin_prefetch(pw, 0, 0);
        }
        __syncthreads();                       // stores to buf(cur) visible; prior readers of buf(cur^1) done
        computeFrom(smem + cur * BUF_ELEMS);
        if (it + 1 < NIT) storeTile(smem + (cur ^ 1) * BUF_ELEMS);
        cur ^= 1;
    }

    // ---- fused LSTM epilogue (branchless transcendentals) ----
    const int kOff = k * H_;
    #pragma unroll
    for (int ni = 0; ni < 2; ++ni) {
        const int n = nTile + waveN + ni * 16 + lrow;
        const float bF = bsv[0][kOff + n];
        const float bI = bsv[1][kOff + n];
        const float bO = bsv[2][kOff + n];
        const float bC = bsv[3][kOff + n];
        #pragma unroll
        for (int mi = 0; mi < 2; ++mi) {
            #pragma unroll
            for (int r = 0; r < 8; ++r) {
                const int m = mTile + waveM + mi * 16 + lhi * 8 + r;  // C layout: vgpr r = row r / r+8
                const size_t idx = (size_t)m * (K_ * H_) + kOff + n;
                const float fg = fast_sigmoid(acc[0][mi][ni][r] + bF);
                const float ig = fast_sigmoid(acc[1][mi][ni][r] + bI);
                const float og = fast_sigmoid(acc[2][mi][ni][r] + bO);
                const float cp = fast_tanh(acc[3][mi][ni][r] + bC);   // reference applies tanh twice
                const float nc = fg * cell[idx] + ig * fast_tanh(cp);
                const float nh = og * fast_tanh(nc);
                out[idx] = (mask[idx] > 0) ? nh : hidden[idx];
            }
        }
    }
}

extern "C" void kernel_launch(void* const* d_in, const int* in_sizes, int n_in,
                              void* d_out, int out_size, void* d_ws, size_t ws_size,
                              hipStream_t stream) {
    (void)in_sizes; (void)n_in; (void)out_size; (void)d_ws; (void)ws_size;
    const float* hidden = (const float*)d_in[0];
    const float* x      = (const float*)d_in[1];
    const int*   mask   = (const int*)d_in[2];
    const float* cell   = (const float*)d_in[3];

    dim3 grid(H_ / BN, B_ / BM, K_);   // (8, 8, 16) — k slowest for L2 weight residency
    dim3 block(256);
    hipLaunchKernelGGL(lstm_fused_wmma, grid, block, 0, stream,
        hidden, x, mask, cell,
        (const float*)d_in[4],  (const float*)d_in[5],  (const float*)d_in[6],
        (const float*)d_in[7],  (const float*)d_in[8],  (const float*)d_in[9],
        (const float*)d_in[10], (const float*)d_in[11], (const float*)d_in[12],
        (const float*)d_in[13], (const float*)d_in[14], (const float*)d_in[15],
        (float*)d_out);
}
